// GRU_base2_23295902613537
// MI455X (gfx1250) — compile-verified, matrix operand order checked by reading, not run on hardware
//
#include <hip/hip_runtime.h>
#include <math.h>

// ---------------------------------------------------------------------------
// Model constants (from reference): B=8, S=128, N=32, E=64, d=300, H=1150
// T=B*S=1024, EE=E+N=96, HEADS=4, OC=75, NG=40000, NS=25000
// ---------------------------------------------------------------------------
#define BB     8
#define SS     128
#define TT     1024
#define NN_    32
#define EEC    96       // E + N self loops
#define DD     300
#define HH     1150
#define HP     1152     // H padded to multiple of 32 for bf16 WMMA K
#define G3     3450     // 3*H
#define G3P    3456     // padded ldc for gate buffer
#define NGC    40000
#define NSC    25000

typedef __attribute__((ext_vector_type(16))) __bf16 v16bf;
typedef __attribute__((ext_vector_type(8)))  float  v8f;

struct FragU { unsigned d[8]; };   // 16 bf16 packed as 8 dwords (32 bytes)

// round-to-nearest-even f32 -> bf16 packers
__device__ __forceinline__ unsigned pkbf2(float a, float b) {
    unsigned ua = __builtin_bit_cast(unsigned, a);
    unsigned ub = __builtin_bit_cast(unsigned, b);
    ua += 0x7FFFu + ((ua >> 16) & 1u);
    ub += 0x7FFFu + ((ub >> 16) & 1u);
    return (ua >> 16) | (ub & 0xFFFF0000u);
}
__device__ __forceinline__ unsigned short bf16of(float a) {
    unsigned u = __builtin_bit_cast(unsigned, a);
    return (unsigned short)((u + 0x7FFFu + ((u >> 16) & 1u)) >> 16);
}

// 8 consecutive f32 (8-byte aligned) -> 4 packed bf16 dwords. Unconditional
// float2 loads => global_load_b64, no exec masking.
__device__ __forceinline__ void load8_cvt(const float* __restrict__ p, unsigned* d) {
    const float2* q = (const float2*)p;
    float2 a = q[0], b = q[1], c = q[2], e = q[3];
    d[0] = pkbf2(a.x, a.y);
    d[1] = pkbf2(b.x, b.y);
    d[2] = pkbf2(c.x, c.y);
    d[3] = pkbf2(e.x, e.y);
}

// Fast path: K-tile fully in bounds; r = row base pointer (row pre-clamped).
// Per ISA 7.12.2 (16-bit A/B 16x32): this lane supplies K columns
// [c0..c0+7] into dwords 0..3 and [c1..c1+7] into dwords 4..7.
__device__ __forceinline__ FragU load_frag_fast(const float* __restrict__ r,
                                                int c0, int c1) {
    FragU f;
    load8_cvt(r + c0, f.d);
    load8_cvt(r + c1, f.d + 4);
    return f;
}

// Tail path: clamp k index (always-in-bounds unconditional load), then
// v_cndmask-select zero for k >= K. No control flow.
__device__ __forceinline__ FragU load_frag_tail(const float* __restrict__ r,
                                                int K, int c0, int c1) {
    float v[16];
#pragma unroll
    for (int i = 0; i < 8; ++i) {
        int k  = c0 + i;
        int kc = k < K ? k : (K - 1);
        float t = r[kc];
        v[i] = (k < K) ? t : 0.0f;
    }
#pragma unroll
    for (int i = 0; i < 8; ++i) {
        int k  = c1 + i;
        int kc = k < K ? k : (K - 1);
        float t = r[kc];
        v[8 + i] = (k < K) ? t : 0.0f;
    }
    FragU f;
#pragma unroll
    for (int i = 0; i < 8; ++i) f.d[i] = pkbf2(v[2 * i], v[2 * i + 1]);
    return f;
}

// ---------------------------------------------------------------------------
// Generic bf16-WMMA GEMM: C[M,N] = A[M,K] * B[N,K]^T (+ bias[N])
// A,B f32 row-major (even ld => 8B alignment), converted to bf16 in regs,
// f32 accumulate. Block = 256 threads = 8 waves tiled 4(M)x2(N) -> 64x32.
// Out-of-range rows are clamped (their products only reach unsaved D slots).
// K-loop unrolled x2; streamed B rows software-prefetched 2 tiles ahead
// (global_prefetch_b8) -- the big weight matrices are touched exactly once.
// ---------------------------------------------------------------------------
__global__ void gemm_bf16(const float* __restrict__ A, int lda,
                          const float* __restrict__ Bw, int ldb,
                          const float* __restrict__ bias,
                          float* __restrict__ C, int ldc,
                          int M, int N, int K) {
    int lane   = threadIdx.x & 31;
    int w      = threadIdx.x >> 5;
    int lane16 = lane & 15;
    int hi     = (lane >> 4) * 8;
    int m0 = blockIdx.y * 64 + (w >> 1) * 16;
    int n0 = blockIdx.x * 32 + (w & 1) * 16;
    long ra = m0 + lane16; if (ra > M - 1) ra = M - 1;   // clamp
    long rb = n0 + lane16; if (rb > N - 1) rb = N - 1;   // clamp
    const float* __restrict__ rA = A  + ra * (long)lda;  // hoisted row bases
    const float* __restrict__ rB = Bw + rb * (long)ldb;
    v8f acc = {};
    int kMain = K & ~31;
    int kDbl  = kMain & ~63;
    int k0 = 0;
    for (; k0 < kDbl; k0 += 64) {
        __builtin_prefetch(rB + k0 + 128, 0, 1);         // 2 k-tiles ahead
        FragU fa0 = load_frag_fast(rA, k0 + hi,      k0 + 16 + hi);
        FragU fb0 = load_frag_fast(rB, k0 + hi,      k0 + 16 + hi);
        FragU fa1 = load_frag_fast(rA, k0 + 32 + hi, k0 + 48 + hi);
        FragU fb1 = load_frag_fast(rB, k0 + 32 + hi, k0 + 48 + hi);
        acc = __builtin_amdgcn_wmma_f32_16x16x32_bf16(
                  false, __builtin_bit_cast(v16bf, fa0),
                  false, __builtin_bit_cast(v16bf, fb0),
                  (short)0, acc, false, false);
        acc = __builtin_amdgcn_wmma_f32_16x16x32_bf16(
                  false, __builtin_bit_cast(v16bf, fa1),
                  false, __builtin_bit_cast(v16bf, fb1),
                  (short)0, acc, false, false);
    }
    if (k0 < kMain) {                                    // one leftover 32-tile
        FragU fa = load_frag_fast(rA, k0 + hi, k0 + 16 + hi);
        FragU fb = load_frag_fast(rB, k0 + hi, k0 + 16 + hi);
        acc = __builtin_amdgcn_wmma_f32_16x16x32_bf16(
                  false, __builtin_bit_cast(v16bf, fa),
                  false, __builtin_bit_cast(v16bf, fb),
                  (short)0, acc, false, false);
    }
    if (kMain < K) {                                     // K % 32 tail
        FragU fa = load_frag_tail(rA, K, kMain + hi, kMain + 16 + hi);
        FragU fb = load_frag_tail(rB, K, kMain + hi, kMain + 16 + hi);
        acc = __builtin_amdgcn_wmma_f32_16x16x32_bf16(
                  false, __builtin_bit_cast(v16bf, fa),
                  false, __builtin_bit_cast(v16bf, fb),
                  (short)0, acc, false, false);
    }
    int m_lo = m0 + ((lane >= 16) ? 8 : 0);
    int n    = n0 + lane16;
    float bv = 0.0f;
    if (bias && n < N) bv = bias[n];
#pragma unroll
    for (int vr = 0; vr < 8; ++vr) {
        int m = m_lo + vr;
        if (m < M && n < N) C[(long)m * ldc + n] = acc[vr] + bv;
    }
}

// ---------------------------------------------------------------------------
// GRU recurrent step: one wave per 16-column tile of H. Computes the three
// gate tiles gh = h @ Whh^T for its columns via WMMA (M=8 batch rows padded
// to 16, K=1152 zero-padded), then gate nonlinearity + h update.
// h is double buffered (bf16 for WMMA A, f32 for the update).
// ---------------------------------------------------------------------------
__global__ void gru_step(const unsigned* __restrict__ hb_in,
                         unsigned short* __restrict__ hb_out,
                         const float* __restrict__ hf_in,
                         float* __restrict__ hf_out,
                         const float* __restrict__ Whh,
                         const float* __restrict__ bhh,
                         const float* __restrict__ Gi,
                         float* __restrict__ xout,
                         float* __restrict__ mainOut, int s) {
    int lane   = threadIdx.x;
    int lane16 = lane & 15;
    int hi     = (lane >> 4) * 8;
    int j0 = blockIdx.x * 16;
    int jb = j0 + lane16;
    bool jok = jb < HH;
    int jc = jok ? jb : (HH - 1);          // clamped column (garbage unsaved)
    const float* __restrict__ w0 = Whh + (long)(0 * HH + jc) * HH;
    const float* __restrict__ w1 = Whh + (long)(1 * HH + jc) * HH;
    const float* __restrict__ w2 = Whh + (long)(2 * HH + jc) * HH;

    v8f acc[3] = {};
    int rbase = lane16 * (HP / 2);         // dword row base into bf16 h [16][1152]
    const int kMain = HH & ~31;            // 1120
    for (int k0 = 0; k0 < kMain; k0 += 32) {
        FragU fa;
        uint4 q0 = *(const uint4*)(hb_in + rbase + ((k0 + hi) >> 1));
        uint4 q1 = *(const uint4*)(hb_in + rbase + ((k0 + 16 + hi) >> 1));
        fa.d[0] = q0.x; fa.d[1] = q0.y; fa.d[2] = q0.z; fa.d[3] = q0.w;
        fa.d[4] = q1.x; fa.d[5] = q1.y; fa.d[6] = q1.z; fa.d[7] = q1.w;
        v16bf av = __builtin_bit_cast(v16bf, fa);
        FragU f0 = load_frag_fast(w0, k0 + hi, k0 + 16 + hi);
        acc[0] = __builtin_amdgcn_wmma_f32_16x16x32_bf16(
                     false, av, false, __builtin_bit_cast(v16bf, f0),
                     (short)0, acc[0], false, false);
        FragU f1 = load_frag_fast(w1, k0 + hi, k0 + 16 + hi);
        acc[1] = __builtin_amdgcn_wmma_f32_16x16x32_bf16(
                     false, av, false, __builtin_bit_cast(v16bf, f1),
                     (short)0, acc[1], false, false);
        FragU f2 = load_frag_fast(w2, k0 + hi, k0 + 16 + hi);
        acc[2] = __builtin_amdgcn_wmma_f32_16x16x32_bf16(
                     false, av, false, __builtin_bit_cast(v16bf, f2),
                     (short)0, acc[2], false, false);
    }
    {   // K tail (k = 1120..1149 real, 1150..1151 zero; A side is zero-padded)
        FragU fa;
        uint4 q0 = *(const uint4*)(hb_in + rbase + ((kMain + hi) >> 1));
        uint4 q1 = *(const uint4*)(hb_in + rbase + ((kMain + 16 + hi) >> 1));
        fa.d[0] = q0.x; fa.d[1] = q0.y; fa.d[2] = q0.z; fa.d[3] = q0.w;
        fa.d[4] = q1.x; fa.d[5] = q1.y; fa.d[6] = q1.z; fa.d[7] = q1.w;
        v16bf av = __builtin_bit_cast(v16bf, fa);
        FragU f0 = load_frag_tail(w0, HH, kMain + hi, kMain + 16 + hi);
        acc[0] = __builtin_amdgcn_wmma_f32_16x16x32_bf16(
                     false, av, false, __builtin_bit_cast(v16bf, f0),
                     (short)0, acc[0], false, false);
        FragU f1 = load_frag_tail(w1, HH, kMain + hi, kMain + 16 + hi);
        acc[1] = __builtin_amdgcn_wmma_f32_16x16x32_bf16(
                     false, av, false, __builtin_bit_cast(v16bf, f1),
                     (short)0, acc[1], false, false);
        FragU f2 = load_frag_tail(w2, HH, kMain + hi, kMain + 16 + hi);
        acc[2] = __builtin_amdgcn_wmma_f32_16x16x32_bf16(
                     false, av, false, __builtin_bit_cast(v16bf, f2),
                     (short)0, acc[2], false, false);
    }
    // lanes 0..15, vgpr vr -> batch row b=vr, column n=lane (lanes 16..31 hold
    // padded M rows 8..15: ignored)
    if (lane < 16 && jok) {
        int j = jb;
#pragma unroll
        for (int b = 0; b < 8; ++b) {
            int row = s * 8 + b;
            const float* gi = Gi + (long)row * G3P;
            float ir = gi[j];
            float iz = gi[HH + j];
            float in = gi[2 * HH + j];
            float hr = acc[0][b] + bhh[j];
            float hz = acc[1][b] + bhh[HH + j];
            float hn = acc[2][b] + bhh[2 * HH + j];
            float r  = 1.0f / (1.0f + expf(-(ir + hr)));
            float z  = 1.0f / (1.0f + expf(-(iz + hz)));
            float nn = tanhf(in + r * hn);
            float hp = hf_in[b * HP + j];
            float h2 = (1.0f - z) * nn + z * hp;
            hf_out[b * HP + j] = h2;
            hb_out[b * HP + j] = bf16of(h2);
            xout[(long)row * HP + j] = h2;
            if (mainOut) mainOut[(long)(b * SS + s) * HP + j] = h2;
        }
    }
}

// ---------------------------------------------------------------------------
// Helper kernels
// ---------------------------------------------------------------------------
__global__ void h_init(const float* __restrict__ h0,
                       float* __restrict__ hf, unsigned short* __restrict__ hb) {
    int idx = blockIdx.x * 256 + threadIdx.x;
    if (idx >= 16 * HP) return;
    int r = idx / HP, c = idx % HP;
    float v = (r < 8 && c < HH) ? h0[r * HH + c] : 0.0f;
    hf[idx] = v;
    hb[idx] = bf16of(v);
}

__global__ void gather_rows(const float* __restrict__ X,
                            const int* __restrict__ idx,
                            float* __restrict__ out) {
    int r = blockIdx.x;
    long src = (long)idx[r] * DD;
    long dst = (long)r * DD;
    for (int f = threadIdx.x; f < DD; f += 128) out[dst + f] = X[src + f];
}

__global__ void transpose300(const float* __restrict__ in, float* __restrict__ out) {
    int i = blockIdx.x * 256 + threadIdx.x;
    if (i >= DD * DD) return;
    int n = i / DD, k = i % DD;
    out[n * DD + k] = in[k * DD + n];
}

// GAT: only node 0's aggregation is consumed, so softmax/aggregate only over
// edges with dst==0 (incl. the (0,0) self loop -> denominator never empty).
__global__ void gat_attn(const float* __restrict__ proj,
                         const int* __restrict__ ei,
                         const float* __restrict__ att_src,
                         const float* __restrict__ att_dst,
                         const float* __restrict__ gat_b,
                         float* __restrict__ sense) {
    int t = blockIdx.x, tid = threadIdx.x;
    __shared__ float asrc[NN_][4];
    __shared__ float adst0[4];
    __shared__ int   ssrc[EEC];
    __shared__ int   sdst[EEC];
    __shared__ float mx[4], dn[4];
    const float* P = proj + (long)t * NN_ * DD;

    if (tid < 128) {                       // a_src for all 32 nodes x 4 heads
        int i = tid >> 2, h = tid & 3;
        const float* pr = P + i * DD + h * 75;
        const float* as = att_src + h * 75;
        float a = 0.0f;
        for (int o = 0; o < 75; ++o) a += pr[o] * as[o];
        asrc[i][h] = a;
    } else if (tid < 132) {                // a_dst for node 0 only
        int h = tid - 128;
        const float* pr = P + h * 75;
        const float* ad = att_dst + h * 75;
        float a = 0.0f;
        for (int o = 0; o < 75; ++o) a += pr[o] * ad[o];
        adst0[h] = a;
    }
    if (tid < EEC) {
        if (tid < 64) { ssrc[tid] = ei[t * 128 + tid]; sdst[tid] = ei[t * 128 + 64 + tid]; }
        else          { ssrc[tid] = tid - 64;          sdst[tid] = tid - 64; }
    }
    __syncthreads();
    if (tid < 4) {                         // per-head max and denom over dst==0
        int h = tid;
        float m = -3.4e38f;
        for (int e = 0; e < EEC; ++e) if (sdst[e] == 0) {
            float v = asrc[ssrc[e]][h] + adst0[h];
            v = v > 0.0f ? v : 0.2f * v;
            m = fmaxf(m, v);
        }
        float sden = 0.0f;
        for (int e = 0; e < EEC; ++e) if (sdst[e] == 0) {
            float v = asrc[ssrc[e]][h] + adst0[h];
            v = v > 0.0f ? v : 0.2f * v;
            sden += expf(v - m);
        }
        mx[h] = m; dn[h] = sden;
    }
    __syncthreads();
    if (tid < DD) {                        // aggregate 300 output features
        int h = tid / 75;
        float a = 0.0f;
        for (int e = 0; e < EEC; ++e) if (sdst[e] == 0) {
            float v = asrc[ssrc[e]][h] + adst0[h];
            v = v > 0.0f ? v : 0.2f * v;
            a += (expf(v - mx[h]) / dn[h]) * P[ssrc[e] * DD + tid];
        }
        sense[(long)t * DD + tid] = a + gat_b[tid];
    }
}

__global__ void build_sig(const float* __restrict__ X,
                          const int* __restrict__ xg,
                          const float* __restrict__ sense,
                          float* __restrict__ sig) {
    int t = blockIdx.x;
    int b = t >> 7, s = t & 127;
    long row  = (long)(s * BB + b) * (2 * DD);
    long xoff = (long)xg[t] * DD;
    for (int f = threadIdx.x; f < 2 * DD; f += 256)
        sig[row + f] = (f < DD) ? X[xoff + f] : sense[(long)t * DD + (f - DD)];
}

__global__ void log_softmax_rows(float* __restrict__ base, int N) {
    int row = blockIdx.x, tid = threadIdx.x;
    float* p = base + (long)row * N;
    __shared__ float red[256];
    float m = -3.4e38f;
    for (int i = tid; i < N; i += 256) m = fmaxf(m, p[i]);
    red[tid] = m; __syncthreads();
    for (int st = 128; st > 0; st >>= 1) {
        if (tid < st) red[tid] = fmaxf(red[tid], red[tid + st]);
        __syncthreads();
    }
    float bm = red[0]; __syncthreads();
    float s = 0.0f;
    for (int i = tid; i < N; i += 256) s += expf(p[i] - bm);
    red[tid] = s; __syncthreads();
    for (int st = 128; st > 0; st >>= 1) {
        if (tid < st) red[tid] += red[tid + st];
        __syncthreads();
    }
    float lg = logf(red[0]);
    for (int i = tid; i < N; i += 256) p[i] = p[i] - bm - lg;
}

// ---------------------------------------------------------------------------
// Host driver. Workspace plan (bytes, needs ~83 MB):
//   XS    @ 0         39321600   (gathered Xs f32 [32768,300]); later aliased:
//                                 GI @ +0 (14.2MB), XA @ +16MB, XB @ +21MB
//   PROJ  @ 39321600  39321600   (proj f32 [32768,300]); later MAINB alias
//   SENSE @ 78643200   1228800
//   SIG   @ 79872000   2457600
//   GATWT @ 82329600    360000
//   HF0/1 @ 82689600   2*73728
//   HB0/1 @ 82837056   2*36864   (end 82910784)
// ---------------------------------------------------------------------------
extern "C" void kernel_launch(void* const* d_in, const int* in_sizes, int n_in,
                              void* d_out, int out_size, void* d_ws, size_t ws_size,
                              hipStream_t stream) {
    const int*   xg      = (const int*)d_in[0];
    const int*   xs      = (const int*)d_in[1];
    const int*   ei      = (const int*)d_in[2];
    const float* X       = (const float*)d_in[3];
    const float* gat_W   = (const float*)d_in[4];
    const float* att_src = (const float*)d_in[5];
    const float* att_dst = (const float*)d_in[6];
    const float* gat_b   = (const float*)d_in[7];
    const float* Wih[5], *Whh[5], *bih[5], *bhh[5];
    for (int i = 0; i < 3; ++i) {        // main layers 0..2
        Wih[i] = (const float*)d_in[8 + 4 * i + 0];
        Whh[i] = (const float*)d_in[8 + 4 * i + 1];
        bih[i] = (const float*)d_in[8 + 4 * i + 2];
        bhh[i] = (const float*)d_in[8 + 4 * i + 3];
    }
    for (int i = 0; i < 2; ++i) {        // sense layers 3..4
        Wih[3 + i] = (const float*)d_in[20 + 4 * i + 0];
        Whh[3 + i] = (const float*)d_in[20 + 4 * i + 1];
        bih[3 + i] = (const float*)d_in[20 + 4 * i + 2];
        bhh[3 + i] = (const float*)d_in[20 + 4 * i + 3];
    }
    const float* lg_W = (const float*)d_in[28];
    const float* lg_b = (const float*)d_in[29];
    const float* ls_W = (const float*)d_in[30];
    const float* ls_b = (const float*)d_in[31];
    const float* h0m  = (const float*)d_in[32];
    const float* h0s  = (const float*)d_in[33];

    char* ws = (char*)d_ws;
    float*          XS    = (float*)(ws + 0);
    float*          PROJ  = (float*)(ws + 39321600);
    float*          SENSE = (float*)(ws + 78643200);
    float*          SIG   = (float*)(ws + 79872000);
    float*          GATWT = (float*)(ws + 82329600);
    float*          HF0   = (float*)(ws + 82689600);
    float*          HF1   = (float*)(ws + 82689600 + 73728);
    unsigned short* HB0   = (unsigned short*)(ws + 82837056);
    unsigned short* HB1   = (unsigned short*)(ws + 82837056 + 36864);
    float*          GI    = (float*)(ws + 0);          // alias XS (after proj)
    float*          XA    = (float*)(ws + 16777216);   // alias XS region
    float*          XB    = (float*)(ws + 22020096);   // alias XS region
    float*          MAINB = (float*)(ws + 39321600);   // alias PROJ (after attn)

    float* outG = (float*)d_out;
    float* outS = outG + (long)TT * NGC;

    // --- front end: gather -> WMMA projection GEMM -> attention -> sig ---
    gather_rows<<<TT * NN_, 128, 0, stream>>>(X, xs, XS);
    transpose300<<<(DD * DD + 255) / 256, 256, 0, stream>>>(gat_W, GATWT);
    gemm_bf16<<<dim3((DD + 31) / 32, (TT * NN_ + 63) / 64), 256, 0, stream>>>(
        XS, DD, GATWT, DD, nullptr, PROJ, DD, TT * NN_, DD, DD);
    gat_attn<<<TT, 320, 0, stream>>>(PROJ, ei, att_src, att_dst, gat_b, SENSE);
    build_sig<<<TT, 256, 0, stream>>>(X, xg, SENSE, SIG);

    // --- GRU layer driver: batched input GEMM + 128 fused recurrent steps ---
    auto run_layer = [&](const float* A, int lda, int K, int li,
                         const float* h0, float* xout, float* mainOut) {
        h_init<<<(16 * HP + 255) / 256, 256, 0, stream>>>(h0, HF0, HB0);
        h_init<<<(16 * HP + 255) / 256, 256, 0, stream>>>(h0, HF1, HB1);
        gemm_bf16<<<dim3((G3 + 31) / 32, (TT + 63) / 64), 256, 0, stream>>>(
            A, lda, Wih[li], K, bih[li], GI, G3P, TT, G3, K);
        for (int s = 0; s < SS; ++s) {
            const unsigned* hbi = (const unsigned*)((s & 1) ? HB1 : HB0);
            unsigned short* hbo = (s & 1) ? HB0 : HB1;
            const float*    hfi = (s & 1) ? HF1 : HF0;
            float*          hfo = (s & 1) ? HF0 : HF1;
            gru_step<<<(HH + 15) / 16, 32, 0, stream>>>(
                hbi, hbo, hfi, hfo, Whh[li], bhh[li], GI, xout, mainOut, s);
        }
    };

    // main stack (3 layers); final layer also writes the permuted [b*S+s] view
    run_layer(SIG, 2 * DD, 2 * DD, 0, h0m + 0 * BB * HH, XA, nullptr);
    run_layer(XA,  HP,     HH,     1, h0m + 1 * BB * HH, XB, nullptr);
    run_layer(XB,  HP,     HH,     2, h0m + 2 * BB * HH, XA, MAINB);
    // globals head -> d_out (in place log-softmax afterwards)
    gemm_bf16<<<dim3((NGC + 31) / 32, (TT + 63) / 64), 256, 0, stream>>>(
        MAINB, HP, lg_W, HH, lg_b, outG, NGC, TT, NGC, HH);

    // senses stack (2 layers); [s*B+b] row order matches reference reshape
    run_layer(SIG, 2 * DD, 2 * DD, 3, h0s + 0 * BB * HH, XB, nullptr);
    run_layer(XB,  HP,     HH,     4, h0s + 1 * BB * HH, XA, nullptr);
    gemm_bf16<<<dim3((NSC + 31) / 32, (TT + 63) / 64), 256, 0, stream>>>(
        XA, HP, ls_W, HH, ls_b, outS, NSC, TT, NSC, HH);

    log_softmax_rows<<<TT, 256, 0, stream>>>(outG, NGC);
    log_softmax_rows<<<TT, 256, 0, stream>>>(outS, NSC);
}
